// RoutedConv2dWithUnlearning_37520834298358
// MI455X (gfx1250) — compile-verified, matrix operand order checked by reading, not run on hardware
//
#include <hip/hip_runtime.h>
#include <hip/hip_bf16.h>

// ---------------- problem constants ----------------
#define B_      32
#define IN_C    256
#define OUT_C   512
#define NF      64          // top-k filters per batch
#define HW      56
#define NPIX    (HW*HW)     // 3136
#define PADW    58
#define PPLANE  (PADW*PADW) // 3364 padded plane
#define KTAPS   9

typedef __attribute__((ext_vector_type(16))) __bf16 bf16x16;
typedef __attribute__((ext_vector_type(8)))  float  f32x8;

union FragAB { uint4 u[2]; bf16x16 v; };
union AccU   { f32x8 v; float f[8]; };

static __device__ __forceinline__ unsigned short f2bf(float f) {
    unsigned int u = __float_as_uint(f);
    unsigned int r = (u + 0x7FFFu + ((u >> 16) & 1u)) >> 16;   // RNE
    return (unsigned short)r;
}

// async global->LDS 16B chunk (CDNA5), GV addressing: 64-bit vgpr address, saddr=off
static __device__ __forceinline__ void async_g2l_b128(unsigned lds_byte_addr,
                                                      const void* gaddr) {
    asm volatile("global_load_async_to_lds_b128 %0, %1, off"
                 :: "v"(lds_byte_addr), "v"((unsigned long long)(uintptr_t)gaddr)
                 : "memory");
}
static __device__ __forceinline__ void wait_asynccnt0() {
    asm volatile("s_wait_asynccnt 0" ::: "memory");
}

// ---------------------------------------------------------------
// 1) prep x: fp32 NCHW -> bf16 channels-last padded [b][py][px][ic]
//    + fused global-avg-pool partial sums (atomicAdd per row)
//    grid = (PADW rows, B_), block = 256 (one thread per ic)
// ---------------------------------------------------------------
__global__ void prep_x_kernel(const float* __restrict__ x,
                              unsigned short* __restrict__ xq,
                              float* __restrict__ pooled) {
    const int py = blockIdx.x, b = blockIdx.y, t = threadIdx.x;
    const bool interior = (py >= 1 && py <= HW);

    __shared__ float tile[IN_C * HW];   // [ic][px], 57344 B

    if (interior) {
        // coalesced along x: 4 channels x 64 lanes per step
        const int icl = t >> 6, px = t & 63;
        if (px < HW) {
            for (int ic0 = 0; ic0 < IN_C; ic0 += 4) {
                tile[(ic0 + icl) * HW + px] =
                    x[((size_t)(b * IN_C + ic0 + icl)) * NPIX + (py - 1) * HW + px];
            }
        }
    }
    __syncthreads();

    if (interior) {                      // per-channel row sum -> pooled accumulator
        float s = 0.f;
        #pragma unroll 8
        for (int px = 0; px < HW; ++px) s += tile[t * HW + px];
        atomicAdd(&pooled[b * IN_C + t], s);
    }

    // coalesced along ic: one 512B run per padded pixel
    unsigned short* dst = xq + ((size_t)(b * PPLANE + py * PADW)) * IN_C;
    for (int px = 0; px < PADW; ++px) {
        float v = (interior && px >= 1 && px <= HW) ? tile[t * HW + (px - 1)] : 0.f;
        dst[(size_t)px * IN_C + t] = f2bf(v);
    }
}

// ---------------------------------------------------------------
// 2) weight repack: fp32 [oc][ic][kh][kw] -> bf16 [khw][oc][ic]
// ---------------------------------------------------------------
__global__ void prep_w_kernel(const float* __restrict__ w,
                              unsigned short* __restrict__ wq) {
    int id = blockIdx.x * blockDim.x + threadIdx.x;          // (khw*OUT_C + oc)*IN_C + ic
    if (id >= KTAPS * OUT_C * IN_C) return;
    int khw = id / (OUT_C * IN_C);
    int rem = id - khw * (OUT_C * IN_C);
    int oc = rem / IN_C, ic = rem - oc * IN_C;
    wq[id] = f2bf(w[((size_t)(oc * IN_C + ic)) * KTAPS + khw]);
}

// ---------------------------------------------------------------
// 3) router: GEMV + exact top-64 rank selection (JAX tie-break: lower idx wins)
//    grid = B_, block = 256
// ---------------------------------------------------------------
__global__ void router_kernel(const float* __restrict__ pooled,
                              const float* __restrict__ rw,
                              const float* __restrict__ rb,
                              int* __restrict__ rank,
                              int* __restrict__ sel) {
    const int b = blockIdx.x, t = threadIdx.x;
    __shared__ float pool[IN_C];
    __shared__ float sc[OUT_C];
    if (t < IN_C) pool[t] = pooled[b * IN_C + t] * (1.f / (float)NPIX); // sums -> mean
    __syncthreads();
    for (int oc = t; oc < OUT_C; oc += 256) {
        const float* wr = rw + (size_t)oc * IN_C;
        float s = rb[oc];
        #pragma unroll 8
        for (int j = 0; j < IN_C; ++j) s += pool[j] * wr[j];
        sc[oc] = s;
    }
    __syncthreads();
    for (int oc = t; oc < OUT_C; oc += 256) {
        float v = sc[oc];
        int cnt = 0;
        for (int j = 0; j < OUT_C; ++j) {
            float o = sc[j];
            cnt += (o > v) || (o == v && j < oc);
        }
        rank[b * OUT_C + oc] = cnt;
        if (cnt < NF) sel[b * NF + cnt] = oc;                // rank-ordered compaction
    }
}

// ---------------------------------------------------------------
// 4) zero-fill non-selected output channels
//    grid = (OUT_C, B_), block = 256
// ---------------------------------------------------------------
__global__ void zero_masked_kernel(const int* __restrict__ rank,
                                   float* __restrict__ out) {
    const int c = blockIdx.x, b = blockIdx.y;
    if (rank[b * OUT_C + c] < NF) return;                    // conv writes these
    float4* o4 = (float4*)(out + ((size_t)(b * OUT_C + c)) * NPIX);
    const float4 z = make_float4(0.f, 0.f, 0.f, 0.f);
    for (int i = threadIdx.x; i < NPIX / 4; i += 256) o4[i] = z;
}

// ---------------------------------------------------------------
// 5) routed implicit-GEMM conv, bf16 WMMA + async global->LDS staging
//    grid = (NPIX/64, B_), block = 128 (4 waves)
//    block tile: M=64 selected channels x N=64 pixels, K = 9*256
// ---------------------------------------------------------------
__global__ void __launch_bounds__(128)
conv_wmma_kernel(const unsigned short* __restrict__ xq,
                 const unsigned short* __restrict__ wq,
                 const float* __restrict__ bias,
                 const int* __restrict__ sel,
                 float* __restrict__ out) {
    const int b  = blockIdx.y;
    const int n0 = blockIdx.x * 64;
    const int t  = threadIdx.x;
    const int wave = t >> 5;            // 0..3 : M group
    const int lane = t & 31;
    const int h    = lane >> 4;         // k-half per ISA 16-bit A/B striping
    const int mn   = lane & 15;         // row (A) / col (B) within 16

    __shared__ __align__(16) unsigned short Atile[64 * 32];  // [oc_l][k]
    __shared__ __align__(16) unsigned short Btile[64 * 32];  // [pix_l][k]
    __shared__ int   sel_l[NF];
    __shared__ float bias_l[NF];

    if (t < NF) {
        int oc = sel[b * NF + t];
        sel_l[t]  = oc;
        bias_l[t] = bias[oc];
    }

    // per-thread staging chunks (16B = 8 bf16): A: c = t, t+128; B: same ids
    // chunk c -> row = c>>2 (oc_l or pix_l), kc = c&3 (8-elem k group)
    const int rA0 = t >> 2, kcA0 = t & 3;
    const int rA1 = (t + 128) >> 2, kcA1 = (t + 128) & 3;

    // precompute per-chunk pixel coords for B
    const int nB0 = n0 + rA0, yB0 = nB0 / HW, xB0 = nB0 - yB0 * HW;
    const int nB1 = n0 + rA1, yB1 = nB1 / HW, xB1 = nB1 - yB1 * HW;

    const unsigned lA0 = (unsigned)(uintptr_t)&Atile[(rA0 * 32 + kcA0 * 8)];
    const unsigned lA1 = (unsigned)(uintptr_t)&Atile[(rA1 * 32 + kcA1 * 8)];
    const unsigned lB0 = (unsigned)(uintptr_t)&Btile[(rA0 * 32 + kcA0 * 8)];
    const unsigned lB1 = (unsigned)(uintptr_t)&Btile[(rA1 * 32 + kcA1 * 8)];

    AccU acc[4];
    #pragma unroll
    for (int i = 0; i < 4; ++i)
        #pragma unroll
        for (int r = 0; r < 8; ++r) acc[i].f[r] = 0.f;

    const unsigned short* xb = xq + ((size_t)b * PPLANE) * IN_C;  // channels-last
    __syncthreads();                                              // sel_l ready

    for (int khw = 0; khw < KTAPS; ++khw) {
        const int kh = khw / 3, kw = khw - kh * 3;
        const unsigned short* wk = wq + (size_t)khw * OUT_C * IN_C;
        const unsigned short* xk0 = xb + ((size_t)((yB0 + kh) * PADW + xB0 + kw)) * IN_C;
        const unsigned short* xk1 = xb + ((size_t)((yB1 + kh) * PADW + xB1 + kw)) * IN_C;

        for (int ic0 = 0; ic0 < IN_C; ic0 += 32) {
            __syncthreads();   // previous fragment reads done
            // A: weights for 64 selected channels, 16B contiguous ic-chunks
            async_g2l_b128(lA0, wk + (size_t)sel_l[rA0] * IN_C + ic0 + kcA0 * 8);
            async_g2l_b128(lA1, wk + (size_t)sel_l[rA1] * IN_C + ic0 + kcA1 * 8);
            // B: channels-last image, 16B contiguous ic-chunks per pixel
            async_g2l_b128(lB0, xk0 + ic0 + kcA0 * 8);
            async_g2l_b128(lB1, xk1 + ic0 + kcA1 * 8);
            wait_asynccnt0();
            __syncthreads();

            // A fragment: row = wave*16+mn, two 16B k-chunks at h*8 and 16+h*8
            FragAB af;
            const unsigned short* ar = &Atile[(wave * 16 + mn) * 32];
            af.u[0] = *(const uint4*)(ar + h * 8);
            af.u[1] = *(const uint4*)(ar + 16 + h * 8);

            #pragma unroll
            for (int nt = 0; nt < 4; ++nt) {
                FragAB bf;
                const unsigned short* br = &Btile[(nt * 16 + mn) * 32];
                bf.u[0] = *(const uint4*)(br + h * 8);
                bf.u[1] = *(const uint4*)(br + 16 + h * 8);
                acc[nt].v = __builtin_amdgcn_wmma_f32_16x16x32_bf16(
                    false, af.v, false, bf.v, (short)0, acc[nt].v, false, false);
            }
        }
    }

    // epilogue: C/D layout — VGPR r, lanes0-15: M=r, lanes16-31: M=8+r; N=lane%16
    #pragma unroll
    for (int nt = 0; nt < 4; ++nt) {
        int n = n0 + nt * 16 + mn;
        #pragma unroll
        for (int r = 0; r < 8; ++r) {
            int oc_l = wave * 16 + r + 8 * h;
            out[((size_t)(b * OUT_C + sel_l[oc_l])) * NPIX + n] = acc[nt].f[r] + bias_l[oc_l];
        }
    }
}

// ---------------------------------------------------------------
extern "C" void kernel_launch(void* const* d_in, const int* in_sizes, int n_in,
                              void* d_out, int out_size, void* d_ws, size_t ws_size,
                              hipStream_t stream) {
    const float* x   = (const float*)d_in[0];
    const float* w   = (const float*)d_in[1];
    const float* bia = (const float*)d_in[2];
    const float* rw  = (const float*)d_in[3];
    const float* rb  = (const float*)d_in[4];
    float* out = (float*)d_out;

    // workspace carve-up (all 256B aligned)
    char* ws = (char*)d_ws;
    size_t off = 0;
    unsigned short* xq = (unsigned short*)(ws + off); off += (size_t)B_ * PPLANE * IN_C * 2;   // 55.1 MB
    unsigned short* wq = (unsigned short*)(ws + off); off += (size_t)KTAPS * OUT_C * IN_C * 2; // 2.4 MB
    float* pooled      = (float*)(ws + off);          off += (size_t)B_ * IN_C * 4;
    int*   rank        = (int*)(ws + off);            off += (size_t)B_ * OUT_C * 4;
    int*   sel         = (int*)(ws + off);            off += (size_t)B_ * NF * 4;
    (void)ws_size; (void)in_sizes; (void)n_in; (void)out_size;

    hipMemsetAsync(pooled, 0, (size_t)B_ * IN_C * 4, stream);   // atomic accumulators
    prep_x_kernel<<<dim3(PADW, B_), 256, 0, stream>>>(x, xq, pooled);
    prep_w_kernel<<<(KTAPS * OUT_C * IN_C + 255) / 256, 256, 0, stream>>>(w, wq);
    router_kernel<<<B_, 256, 0, stream>>>(pooled, rw, rb, rank, sel);
    zero_masked_kernel<<<dim3(OUT_C, B_), 256, 0, stream>>>(rank, out);
    conv_wmma_kernel<<<dim3(NPIX / 64, B_), 128, 0, stream>>>(xq, wq, bia, sel, out);
}
